// CLIPAttention_58643483460011
// MI455X (gfx1250) — matrix operand order repaired and checked by reading, not compiled, not run on hardware
//
#include <hip/hip_runtime.h>

// ---------------------------------------------------------------------------
// CLIPAttention forward for MI455X (gfx1250, wave32, WMMA f32_16x16x32_f16)
//   - 32x64 output tile per wave in the dense GEMMs (1.5 b128 loads / WMMA)
//   - attention K/V tiles staged in LDS once per block (4x traffic cut)
// ---------------------------------------------------------------------------

typedef __attribute__((ext_vector_type(16))) _Float16 v16h;
typedef __attribute__((ext_vector_type(8)))  float    v8f;

#define HIDDEN   1024
#define HEADS    16
#define HEAD_DIM 64
#define SEQ      1024
#define BATCH    8
#define NROWS    (BATCH * SEQ)          // 8192
#define ATT_SCALE 0.125f                // 1/sqrt(64)

union FragU { uint4 u[2]; v16h h; };

// A-matrix fragment (16x32 f16), per ISA 7.12.2:
//   lane l: m = l%16, g = l/16 ; VGPR0..3 hold K = g*8 + {0..7},
//   VGPR4..7 hold K = 16 + g*8 + {0..7}  -> two 16B loads.
__device__ __forceinline__ v16h load_frag_A(const _Float16* p, int ld, int row,
                                            int kb, int g) {
  const _Float16* q = p + (size_t)row * ld + kb + g * 8;
  FragU f;
  f.u[0] = *reinterpret_cast<const uint4*>(q);
  f.u[1] = *reinterpret_cast<const uint4*>(q + 16);
  return f.h;
}

// B-matrix fragment (32x16 f16) loaded from B^T storage (row n, K contiguous):
//   lane l: n = l%16, g = l/16 ; VGPR0..3 hold K = g*8 + {0..7},
//   VGPR4..7 hold K = 16 + g*8 + {0..7}.
__device__ __forceinline__ v16h load_frag_B(const _Float16* p, int ld, int row,
                                            int kb, int g) {
  const _Float16* q = p + (size_t)row * ld + kb + g * 8;
  FragU f;
  f.u[0] = *reinterpret_cast<const uint4*>(q);
  f.u[1] = *reinterpret_cast<const uint4*>(q + 16);
  return f.h;
}

__device__ __forceinline__ v8f wmma_f16(v16h a, v16h b, v8f c) {
  return __builtin_amdgcn_wmma_f32_16x16x32_f16(false, a, false, b,
                                                (short)0, c, false, false);
}

// ---------------------------------------------------------------------------
// fp32 -> f16 conversion, 4 elements per thread
// ---------------------------------------------------------------------------
__global__ void cvt_f32_f16(const float* __restrict__ src,
                            _Float16* __restrict__ dst, int n4) {
  int i = blockIdx.x * blockDim.x + threadIdx.x;
  if (i < n4) {
    float4 v = reinterpret_cast<const float4*>(src)[i];
    union { _Float16 h[4]; uint2 u; } o;
    o.h[0] = (_Float16)v.x; o.h[1] = (_Float16)v.y;
    o.h[2] = (_Float16)v.z; o.h[3] = (_Float16)v.w;
    reinterpret_cast<uint2*>(dst)[i] = o.u;
  }
}

// ---------------------------------------------------------------------------
// Fused Q/K/V projections.  grid=(128 rowBlocks, 8 colBlocks, 3={q,k,v}),
// block=128 (4 waves in 2x2).  Each wave computes a 32x64 tile of
// (X @ W^T + b): 2 A-frags x 4 B-frags -> 8 WMMAs per K-chunk.
// Q stored scaled, [b][h][s][d]; K stored [b][h][s][d]; V stored [b][h][d][s].
// ---------------------------------------------------------------------------
__global__ __launch_bounds__(128) void qkv_proj_kernel(
    const _Float16* __restrict__ X,
    const _Float16* __restrict__ Wq, const float* __restrict__ bq,
    const _Float16* __restrict__ Wk, const float* __restrict__ bk,
    const _Float16* __restrict__ Wv, const float* __restrict__ bv,
    _Float16* __restrict__ Q, _Float16* __restrict__ K,
    _Float16* __restrict__ Vt) {
  const int lane = threadIdx.x & 31;
  const int wv   = threadIdx.x >> 5;
  const int g    = lane >> 4;
  const int n16  = lane & 15;
  const int mi   = wv >> 1;                       // 0..1
  const int ni   = wv & 1;                        // 0..1
  const int rowBase = blockIdx.x * 64 + mi * 32;  // 32 rows per wave
  const int colBase = blockIdx.y * 128 + ni * 64; // 64 cols per wave
  const int mode    = blockIdx.z;                 // 0=Q 1=K 2=V

  const _Float16* Wm = (mode == 0) ? Wq : (mode == 1) ? Wk : Wv;
  const float*    bm = (mode == 0) ? bq : (mode == 1) ? bk : bv;

  v8f acc[2][4];
#pragma unroll
  for (int i = 0; i < 2; ++i)
#pragma unroll
    for (int t = 0; t < 4; ++t) acc[i][t] = (v8f){};

  for (int kb = 0; kb < HIDDEN; kb += 32) {
    v16h a0 = load_frag_A(X, HIDDEN, rowBase + n16,      kb, g);
    v16h a1 = load_frag_A(X, HIDDEN, rowBase + 16 + n16, kb, g);
    v16h b0 = load_frag_B(Wm, HIDDEN, colBase + 0  + n16, kb, g);
    v16h b1 = load_frag_B(Wm, HIDDEN, colBase + 16 + n16, kb, g);
    v16h b2 = load_frag_B(Wm, HIDDEN, colBase + 32 + n16, kb, g);
    v16h b3 = load_frag_B(Wm, HIDDEN, colBase + 48 + n16, kb, g);
    acc[0][0] = wmma_f16(a0, b0, acc[0][0]);
    acc[0][1] = wmma_f16(a0, b1, acc[0][1]);
    acc[0][2] = wmma_f16(a0, b2, acc[0][2]);
    acc[0][3] = wmma_f16(a0, b3, acc[0][3]);
    acc[1][0] = wmma_f16(a1, b0, acc[1][0]);
    acc[1][1] = wmma_f16(a1, b1, acc[1][1]);
    acc[1][2] = wmma_f16(a1, b2, acc[1][2]);
    acc[1][3] = wmma_f16(a1, b3, acc[1][3]);
  }

  const float scl = (mode == 0) ? ATT_SCALE : 1.0f;
#pragma unroll
  for (int t = 0; t < 4; ++t) {
    const int   n    = colBase + t * 16 + n16;
    const float bias = bm[n];
    const int   hh   = n >> 6;
    const int   d    = n & 63;
#pragma unroll
    for (int i = 0; i < 2; ++i) {
#pragma unroll
      for (int r = 0; r < 8; ++r) {
        const int R  = rowBase + i * 16 + r + 8 * g;   // C-layout: M = r + 8g
        const int b_ = R >> 10;
        const int s  = R & (SEQ - 1);
        const _Float16 hv = (_Float16)((acc[i][t][r] + bias) * scl);
        if (mode == 0)
          Q[((size_t)((b_ * HEADS + hh) * SEQ + s)) * HEAD_DIM + d] = hv;
        else if (mode == 1)
          K[((size_t)((b_ * HEADS + hh) * SEQ + s)) * HEAD_DIM + d] = hv;
        else
          Vt[((size_t)((b_ * HEADS + hh) * HEAD_DIM + d)) * SEQ + s] = hv;
      }
    }
  }
}

// ---------------------------------------------------------------------------
// Flash-style attention.  grid=(16 qTiles, 16 heads, 8 batch), block=128.
// Block cooperatively stages the 32-key K/V tiles in LDS (they are identical
// for all 4 waves); each wave owns 16 query rows.
// ---------------------------------------------------------------------------
__global__ __launch_bounds__(128) void attn_kernel(
    const _Float16* __restrict__ Q, const _Float16* __restrict__ K,
    const _Float16* __restrict__ Vt, _Float16* __restrict__ attn) {
  __shared__ __align__(16) _Float16 Ks[32 * HEAD_DIM];   // [key_local][d], 4KB
  __shared__ __align__(16) _Float16 Vs[HEAD_DIM * 32];   // [d][key_local], 4KB
  __shared__ __align__(16) _Float16 Pbuf[4][16 * 32];    // per-wave P tile

  const int tid  = threadIdx.x;
  const int lane = tid & 31;
  const int wv   = tid >> 5;
  const int g    = lane >> 4;
  const int n16  = lane & 15;
  const int b    = blockIdx.z;
  const int hh   = blockIdx.y;
  const int qbase = blockIdx.x * 64 + wv * 16;

  const _Float16* Qp = Q  + ((size_t)(b * HEADS + hh) * SEQ) * HEAD_DIM;
  const _Float16* Kp = K  + ((size_t)(b * HEADS + hh) * SEQ) * HEAD_DIM;
  const _Float16* Vp = Vt + ((size_t)(b * HEADS + hh) * HEAD_DIM) * SEQ;

  // Q fragments (held for the whole key loop): d=0..31 and d=32..63
  const v16h qa0 = load_frag_A(Qp, HEAD_DIM, qbase + n16, 0,  g);
  const v16h qa1 = load_frag_A(Qp, HEAD_DIM, qbase + n16, 32, g);

  v8f o0 = {}, o1 = {}, o2 = {}, o3 = {};
  float mrun[8], lrun[8];
#pragma unroll
  for (int r = 0; r < 8; ++r) { mrun[r] = -__builtin_inff(); lrun[r] = 0.0f; }

  _Float16* P = &Pbuf[wv][0];

  for (int jb = 0; jb < SEQ; jb += 32) {
    // ---- cooperative stage of K tile (32x64) and V tile (64x32) ----
    {
      // K tile: 256 uint4 chunks, 2 per thread
      const int i0 = tid;          // uint4 index
      const int i1 = tid + 128;
      const int kr0 = i0 >> 3, kc0 = (i0 & 7) * 8;
      const int kr1 = i1 >> 3, kc1 = (i1 & 7) * 8;
      reinterpret_cast<uint4*>(Ks)[i0] =
          *reinterpret_cast<const uint4*>(Kp + (size_t)(jb + kr0) * HEAD_DIM + kc0);
      reinterpret_cast<uint4*>(Ks)[i1] =
          *reinterpret_cast<const uint4*>(Kp + (size_t)(jb + kr1) * HEAD_DIM + kc1);
      // V tile: rows d (64), 32 keys each -> 256 uint4 chunks
      const int vr0 = i0 >> 2, vc0 = (i0 & 3) * 8;
      const int vr1 = i1 >> 2, vc1 = (i1 & 3) * 8;
      reinterpret_cast<uint4*>(Vs)[i0] =
          *reinterpret_cast<const uint4*>(Vp + (size_t)vr0 * SEQ + jb + vc0);
      reinterpret_cast<uint4*>(Vs)[i1] =
          *reinterpret_cast<const uint4*>(Vp + (size_t)vr1 * SEQ + jb + vc1);
    }
    __syncthreads();

    // ---- S = Q K^T for 32 keys (two 16x16 f32 tiles) from LDS ----
    v8f s0 = {}, s1 = {};
    {
      v16h k00 = load_frag_B(Ks, HEAD_DIM, n16,      0,  g);
      v16h k01 = load_frag_B(Ks, HEAD_DIM, n16,      32, g);
      s0 = wmma_f16(qa0, k00, s0);
      s0 = wmma_f16(qa1, k01, s0);
      v16h k10 = load_frag_B(Ks, HEAD_DIM, 16 + n16, 0,  g);
      v16h k11 = load_frag_B(Ks, HEAD_DIM, 16 + n16, 32, g);
      s1 = wmma_f16(qa0, k10, s1);
      s1 = wmma_f16(qa1, k11, s1);
    }

    // ---- online softmax (row = r + 8g spans one 16-lane half) ----
    float p0[8], p1[8];
#pragma unroll
    for (int r = 0; r < 8; ++r) {
      float hm = fmaxf(s0[r], s1[r]);
      hm = fmaxf(hm, __shfl_xor(hm, 1, 16));
      hm = fmaxf(hm, __shfl_xor(hm, 2, 16));
      hm = fmaxf(hm, __shfl_xor(hm, 4, 16));
      hm = fmaxf(hm, __shfl_xor(hm, 8, 16));
      const float mnew = fmaxf(mrun[r], hm);
      const float corr = __expf(mrun[r] - mnew);
      p0[r] = __expf(s0[r] - mnew);
      p1[r] = __expf(s1[r] - mnew);
      float rs = p0[r] + p1[r];
      rs += __shfl_xor(rs, 1, 16);
      rs += __shfl_xor(rs, 2, 16);
      rs += __shfl_xor(rs, 4, 16);
      rs += __shfl_xor(rs, 8, 16);
      lrun[r] = lrun[r] * corr + rs;
      mrun[r] = mnew;
      o0[r] *= corr; o1[r] *= corr; o2[r] *= corr; o3[r] *= corr;
    }

    // ---- transpose P (C-layout -> A-layout) through private LDS tile ----
#pragma unroll
    for (int r = 0; r < 8; ++r) {
      const int m = r + 8 * g;
      P[m * 32 + n16]      = (_Float16)p0[r];
      P[m * 32 + 16 + n16] = (_Float16)p1[r];
    }
    __syncthreads();
    const v16h pa = load_frag_A(P, 32, n16, 0, g);

    // ---- O += P V (V tile in LDS: rows = d, K = keys) ----
    v16h v0 = load_frag_B(Vs, 32, 0 * 16 + n16, 0, g);
    v16h v1 = load_frag_B(Vs, 32, 1 * 16 + n16, 0, g);
    v16h v2 = load_frag_B(Vs, 32, 2 * 16 + n16, 0, g);
    v16h v3 = load_frag_B(Vs, 32, 3 * 16 + n16, 0, g);
    o0 = wmma_f16(pa, v0, o0);
    o1 = wmma_f16(pa, v1, o1);
    o2 = wmma_f16(pa, v2, o2);
    o3 = wmma_f16(pa, v3, o3);
    __syncthreads();   // Ks/Vs/P consumed before next iteration overwrites
  }

  // ---- normalize and store [b][s][h*64+d] (f16) ----
#pragma unroll
  for (int r = 0; r < 8; ++r) {
    const float inv = 1.0f / lrun[r];
    const int   q   = qbase + r + 8 * g;
    const size_t ro = ((size_t)(b * SEQ + q)) * HIDDEN + hh * HEAD_DIM;
    attn[ro + 0 * 16 + n16] = (_Float16)(o0[r] * inv);
    attn[ro + 1 * 16 + n16] = (_Float16)(o1[r] * inv);
    attn[ro + 2 * 16 + n16] = (_Float16)(o2[r] * inv);
    attn[ro + 3 * 16 + n16] = (_Float16)(o3[r] * inv);
  }
}

// ---------------------------------------------------------------------------
// Output projection: out = attn @ Wo^T + bo, fp32 result.
// grid=(128, 8), block=128; 32x64 tile per wave (2x2 waves per block).
// ---------------------------------------------------------------------------
__global__ __launch_bounds__(128) void out_proj_kernel(
    const _Float16* __restrict__ A, const _Float16* __restrict__ Wo,
    const float* __restrict__ bo, float* __restrict__ out) {
  const int lane = threadIdx.x & 31;
  const int wv   = threadIdx.x >> 5;
  const int g    = lane >> 4;
  const int n16  = lane & 15;
  const int mi   = wv >> 1;
  const int ni   = wv & 1;
  const int rowBase = blockIdx.x * 64 + mi * 32;
  const int colBase = blockIdx.y * 128 + ni * 64;

  v8f acc[2][4];
#pragma unroll
  for (int i = 0; i < 2; ++i)
#pragma unroll
    for (int t = 0; t < 4; ++t) acc[i][t] = (v8f){};

  for (int kb = 0; kb < HIDDEN; kb += 32) {
    v16h a0 = load_frag_A(A, HIDDEN, rowBase + n16,      kb, g);
    v16h a1 = load_frag_A(A, HIDDEN, rowBase + 16 + n16, kb, g);
    v16h b0 = load_frag_B(Wo, HIDDEN, colBase + 0  + n16, kb, g);
    v16h b1 = load_frag_B(Wo, HIDDEN, colBase + 16 + n16, kb, g);
    v16h b2 = load_frag_B(Wo, HIDDEN, colBase + 32 + n16, kb, g);
    v16h b3 = load_frag_B(Wo, HIDDEN, colBase + 48 + n16, kb, g);
    acc[0][0] = wmma_f16(a0, b0, acc[0][0]);
    acc[0][1] = wmma_f16(a0, b1, acc[0][1]);
    acc[0][2] = wmma_f16(a0, b2, acc[0][2]);
    acc[0][3] = wmma_f16(a0, b3, acc[0][3]);
    acc[1][0] = wmma_f16(a1, b0, acc[1][0]);
    acc[1][1] = wmma_f16(a1, b1, acc[1][1]);
    acc[1][2] = wmma_f16(a1, b2, acc[1][2]);
    acc[1][3] = wmma_f16(a1, b3, acc[1][3]);
  }

#pragma unroll
  for (int t = 0; t < 4; ++t) {
    const int   n    = colBase + t * 16 + n16;
    const float bias = bo[n];
#pragma unroll
    for (int i = 0; i < 2; ++i) {
#pragma unroll
      for (int r = 0; r < 8; ++r) {
        const int R = rowBase + i * 16 + r + 8 * g;
        out[(size_t)R * HIDDEN + n] = acc[i][t][r] + bias;
      }
    }
  }
}

// ---------------------------------------------------------------------------
// Host launcher.  Workspace layout (needs 88 MB):
//   [0,16M)   X f16            [16M,18M) Wq f16   [18M,20M) Wk f16
//   [20M,22M) Wv f16           [22M,24M) Wo f16
//   [24M,40M) Q  [b][h][s][d]  [40M,56M) K [b][h][s][d]
//   [56M,72M) V^T [b][h][d][s] [72M,88M) attn f16 [b][s][hidden]
// ---------------------------------------------------------------------------
extern "C" void kernel_launch(void* const* d_in, const int* in_sizes, int n_in,
                              void* d_out, int out_size, void* d_ws,
                              size_t ws_size, hipStream_t stream) {
  (void)in_sizes; (void)n_in; (void)out_size; (void)ws_size;
  const float* hs = (const float*)d_in[0];
  const float* Wq = (const float*)d_in[1];
  const float* bq = (const float*)d_in[2];
  const float* Wk = (const float*)d_in[3];
  const float* bk = (const float*)d_in[4];
  const float* Wv = (const float*)d_in[5];
  const float* bv = (const float*)d_in[6];
  const float* Wo = (const float*)d_in[7];
  const float* bo = (const float*)d_in[8];

  char* ws = (char*)d_ws;
  _Float16* Xh  = (_Float16*)(ws + ((size_t)0  << 20));
  _Float16* Wqh = (_Float16*)(ws + ((size_t)16 << 20));
  _Float16* Wkh = (_Float16*)(ws + ((size_t)18 << 20));
  _Float16* Wvh = (_Float16*)(ws + ((size_t)20 << 20));
  _Float16* Woh = (_Float16*)(ws + ((size_t)22 << 20));
  _Float16* Qh  = (_Float16*)(ws + ((size_t)24 << 20));
  _Float16* Kh  = (_Float16*)(ws + ((size_t)40 << 20));
  _Float16* Vth = (_Float16*)(ws + ((size_t)56 << 20));
  _Float16* Ah  = (_Float16*)(ws + ((size_t)72 << 20));

  const int n4x = NROWS * HIDDEN / 4;     // 2M vec4 elements
  const int n4w = HIDDEN * HIDDEN / 4;    // 256K vec4 elements
  cvt_f32_f16<<<(n4x + 255) / 256, 256, 0, stream>>>(hs, Xh, n4x);
  cvt_f32_f16<<<(n4w + 255) / 256, 256, 0, stream>>>(Wq, Wqh, n4w);
  cvt_f32_f16<<<(n4w + 255) / 256, 256, 0, stream>>>(Wk, Wkh, n4w);
  cvt_f32_f16<<<(n4w + 255) / 256, 256, 0, stream>>>(Wv, Wvh, n4w);
  cvt_f32_f16<<<(n4w + 255) / 256, 256, 0, stream>>>(Wo, Woh, n4w);

  qkv_proj_kernel<<<dim3(NROWS / 64, HIDDEN / 128, 3), 128, 0, stream>>>(
      Xh, Wqh, bq, Wkh, bk, Wvh, bv, Qh, Kh, Vth);

  attn_kernel<<<dim3(SEQ / 64, HEADS, BATCH), 128, 0, stream>>>(Qh, Kh, Vth,
                                                                Ah);

  out_proj_kernel<<<dim3(NROWS / 64, HIDDEN / 128), 128, 0, stream>>>(
      Ah, Woh, bo, (float*)d_out);
}
